// MultiHeadSelfAttention_39573828665384
// MI455X (gfx1250) — compile-verified, matrix operand order checked by reading, not run on hardware
//
#include <hip/hip_runtime.h>

#define EDIM   1024
#define NHEADS 16
#define HDIM   64
#define BATCH  4
#define SEQ    1024
#define MTOT   (BATCH * SEQ)   // 4096
#define RBLK   128             // Q rows per attention workgroup
#define PSTR   1032            // padded LDS row stride (halves): conflict-free frags

typedef _Float16 v16h __attribute__((ext_vector_type(16)));
typedef _Float16 v8h  __attribute__((ext_vector_type(8)));
typedef float    v8f  __attribute__((ext_vector_type(8)));

union FragU { v16h v; v8h h[2]; };

// Load a 16x32 f16 A/B fragment row slice for this lane.
// rowk points at &Mat[row][k0]; hi = (lane>>4)*8 selects the K sub-halves
// per the CDNA5 16-bit A-matrix layout: lanes 0-15 hold K {0..7,16..23},
// lanes 16-31 hold K {8..15,24..31}.
__device__ __forceinline__ v16h load_frag(const _Float16* rowk, int hi) {
    FragU f;
    f.h[0] = *reinterpret_cast<const v8h*>(rowk + hi);
    f.h[1] = *reinterpret_cast<const v8h*>(rowk + 16 + hi);
    return f.v;
}

__device__ __forceinline__ v8f wmma16(v16h a, v16h b, v8f c) {
    return __builtin_amdgcn_wmma_f32_16x16x32_f16(
        /*neg_a=*/false, a, /*neg_b=*/false, b,
        /*c_mod=*/(short)0, c, /*reuse_a=*/false, /*reuse_b=*/false);
}

// ---------------------------------------------------------------------------
// f32 -> f16 conversion
// ---------------------------------------------------------------------------
__global__ void mhsa_cvt_f16(const float* __restrict__ in,
                             _Float16* __restrict__ out, int n) {
    int i = blockIdx.x * blockDim.x + threadIdx.x;
    if (i < n) out[i] = (_Float16)in[i];
}

// ---------------------------------------------------------------------------
// GEMM helpers: 6-fragment load set, 8-WMMA compute set
// ---------------------------------------------------------------------------
__device__ __forceinline__ void load6(v16h (&AA)[4], v16h (&BB)[2],
                                      const _Float16* const (&ar)[4],
                                      const _Float16* const (&br)[2],
                                      int k0, int hi) {
#pragma unroll
    for (int i = 0; i < 4; ++i) AA[i] = load_frag(ar[i] + k0, hi);
#pragma unroll
    for (int j = 0; j < 2; ++j) BB[j] = load_frag(br[j] + k0, hi);
}

__device__ __forceinline__ void mma8(v8f (&acc)[4][2],
                                     const v16h (&AA)[4],
                                     const v16h (&BB)[2]) {
#pragma unroll
    for (int i = 0; i < 4; ++i)
#pragma unroll
        for (int j = 0; j < 2; ++j)
            acc[i][j] = wmma16(AA[i], BB[j], acc[i][j]);
}

// ---------------------------------------------------------------------------
// C = A(MxK) @ W(NxK)^T + bias   (both operands row-major f16; K % 64 == 32)
// mode 0: f16 row-major C (MxN)
// mode 1: f16 head-transposed V: VT[b*EDIM*SEQ + n*SEQ + p], m = b*SEQ + p
// mode 2: f32 row-major C (MxN)
// Block: 256 threads = 8 waves (2x4); block tile 128x128;
// wave tile 64x32 = 4x2 WMMA accumulators; ping-pong double-buffered k-loop
// (no loop-carried copies -> no mov/nop hazard overhead).
// ---------------------------------------------------------------------------
__global__ __launch_bounds__(256)
void mhsa_gemm_xwt(const _Float16* __restrict__ A,
                   const _Float16* __restrict__ W,
                   const float* __restrict__ bias,
                   void* __restrict__ Cout,
                   int M, int N, int K, int mode) {
    const int lane = threadIdx.x & 31;
    const int wid  = threadIdx.x >> 5;
    const int lr   = lane & 15;
    const int hi   = (lane >> 4) * 8;

    const int wm = wid >> 2;                       // 0..1 (64-row slabs)
    const int wn = wid & 3;                        // 0..3 (32-col slabs)
    const int m0 = blockIdx.y * 128 + wm * 64;
    const int n0 = blockIdx.x * 128 + wn * 32;

    const _Float16* ar[4];
    const _Float16* br[2];
#pragma unroll
    for (int i = 0; i < 4; ++i)
        ar[i] = A + (size_t)(m0 + i * 16 + lr) * K;
#pragma unroll
    for (int j = 0; j < 2; ++j)
        br[j] = W + (size_t)(n0 + j * 16 + lr) * K;

    v8f acc[4][2] = {};
    v16h a0[4], b0[2], a1[4], b1[2];

    load6(a0, b0, ar, br, 0, hi);
#pragma unroll 1
    for (int k0 = 32; k0 + 64 <= K; k0 += 64) {
        load6(a1, b1, ar, br, k0, hi);       // step k0 -> buffer 1
        mma8(acc, a0, b0);                   // compute step k0-32
        load6(a0, b0, ar, br, k0 + 32, hi);  // step k0+32 -> buffer 0
        mma8(acc, a1, b1);                   // compute step k0
    }
    mma8(acc, a0, b0);                       // final step (K-32)

#pragma unroll
    for (int i = 0; i < 4; ++i) {
#pragma unroll
        for (int j = 0; j < 2; ++j) {
            const int col     = n0 + j * 16 + lr;       // C column (lane)
            const int rowbase = m0 + i * 16 + hi;       // C rows rowbase..+7
            const float b = bias ? bias[col] : 0.0f;
            if (mode == 0) {
                _Float16* O = (_Float16*)Cout;
#pragma unroll
                for (int e = 0; e < 8; ++e)
                    O[(size_t)(rowbase + e) * N + col] =
                        (_Float16)(acc[i][j][e] + b);
            } else if (mode == 1) {
                _Float16* VT = (_Float16*)Cout;
                const int bb = rowbase >> 10;           // batch index
                const int p0 = rowbase & 1023;          // seq pos base
                v8h o;
#pragma unroll
                for (int e = 0; e < 8; ++e)
                    o[e] = (_Float16)(acc[i][j][e] + b);
                *reinterpret_cast<v8h*>(
                    VT + (size_t)bb * EDIM * SEQ + (size_t)col * SEQ + p0) = o;
            } else {
                float* O = (float*)Cout;
#pragma unroll
                for (int e = 0; e < 8; ++e)
                    O[(size_t)(rowbase + e) * N + col] = acc[i][j][e] + b;
            }
        }
    }
}

// ---------------------------------------------------------------------------
// Attention: one block per (b*16+a, 128-row Q block). 256 threads = 8 waves.
// LDS: S = 128 x PSTR f16 (~258KB <= 320KB WGP LDS); scores then (in-place)
// softmax probabilities. 128-row blocks cut K/V L2 re-reads to 8x per head.
// ---------------------------------------------------------------------------
__global__ __launch_bounds__(256)
void mhsa_attn(const _Float16* __restrict__ Qh,
               const _Float16* __restrict__ Kh,
               const _Float16* __restrict__ VT,
               _Float16* __restrict__ Ctx) {
    extern __shared__ char smem[];
    _Float16* S = (_Float16*)smem;     // [RBLK][PSTR]

    const int ba = blockIdx.x;
    const int b  = ba >> 4;
    const int a  = ba & 15;
    const int r0 = blockIdx.y * RBLK;

    const int lane = threadIdx.x & 31;
    const int wid  = threadIdx.x >> 5;
    const int lr   = lane & 15;
    const int hi   = (lane >> 4) * 8;

    const size_t rowB    = (size_t)b * SEQ;   // global row base for Q/K
    const int    colBase = a * HDIM;          // column offset within EDIM

    // --- Phase 1: S = scale * Q Kt ------------------------------------------
    v16h qf[8][2];                            // resident Q fragments (8 Mtiles)
#pragma unroll
    for (int mt = 0; mt < 8; ++mt)
#pragma unroll
        for (int ks = 0; ks < 2; ++ks)
            qf[mt][ks] = load_frag(
                Qh + (rowB + r0 + mt * 16 + lr) * EDIM + colBase + ks * 32, hi);

    const float scale = 0.125f;  // 1/sqrt(HDIM)
#pragma unroll 1
    for (int t = 0; t < 8; ++t) {
        const int nt   = wid * 8 + t;        // 0..63: 16-col tile of keys
        const int jrow = nt * 16 + lr;       // key index this lane loads
        const _Float16* krow = Kh + (rowB + jrow) * EDIM + colBase;
        v16h kf0 = load_frag(krow, hi);
        v16h kf1 = load_frag(krow + 32, hi);
#pragma unroll
        for (int mt = 0; mt < 8; ++mt) {
            v8f acc = {};
            acc = wmma16(qf[mt][0], kf0, acc);
            acc = wmma16(qf[mt][1], kf1, acc);
            const int col  = nt * 16 + lr;
            const int rowb = mt * 16 + hi;
#pragma unroll
            for (int e = 0; e < 8; ++e)
                S[(rowb + e) * PSTR + col] = (_Float16)(acc[e] * scale);
        }
    }
    __syncthreads();

    // --- Phase 2: exact row softmax, in place (wave handles 16 rows) --------
#pragma unroll 1
    for (int rr = 0; rr < 16; ++rr) {
        const int r = wid * 16 + rr;
        _Float16* Srow = S + r * PSTR;
        float vals[32];
        float m = -3.0e38f;
#pragma unroll
        for (int idx = 0; idx < 32; ++idx) {
            float t = (float)Srow[idx * 32 + lane];
            vals[idx] = t;
            m = fmaxf(m, t);
        }
#pragma unroll
        for (int off = 16; off > 0; off >>= 1)
            m = fmaxf(m, __shfl_xor(m, off, 32));
        float s = 0.0f;
#pragma unroll
        for (int idx = 0; idx < 32; ++idx) {
            vals[idx] = __expf(vals[idx] - m);
            s += vals[idx];
        }
#pragma unroll
        for (int off = 16; off > 0; off >>= 1)
            s += __shfl_xor(s, off, 32);
        const float inv = 1.0f / s;
#pragma unroll
        for (int idx = 0; idx < 32; ++idx)
            Srow[idx * 32 + lane] = (_Float16)(vals[idx] * inv);
    }
    __syncthreads();

    // --- Phase 3: ctx = P @ V (4 output tiles per wave; V frag reused 4x) ---
    const int dt  = wid & 3;          // 16-col slab of head dim
    const int mtg = (wid >> 2) * 4;   // first of 4 row tiles
    const _Float16* vrow = VT + (size_t)b * EDIM * SEQ +
                           (size_t)(colBase + dt * 16 + lr) * SEQ;
    v8f acc[4] = {};
#pragma unroll 2
    for (int j0 = 0; j0 < SEQ; j0 += 32) {
        v16h vf = load_frag(vrow + j0, hi);   // global (VT row-major)
#pragma unroll
        for (int q = 0; q < 4; ++q) {
            const _Float16* prow = S + ((mtg + q) * 16 + lr) * PSTR;
            v16h pf = load_frag(prow + j0, hi);   // LDS, conflict-free
            acc[q] = wmma16(pf, vf, acc[q]);
        }
    }
    const int col = colBase + dt * 16 + lr;
#pragma unroll
    for (int q = 0; q < 4; ++q) {
        const int rowb = r0 + (mtg + q) * 16 + hi;
#pragma unroll
        for (int e = 0; e < 8; ++e)
            Ctx[(rowB + rowb + e) * EDIM + col] = (_Float16)acc[q][e];
    }
}

// ---------------------------------------------------------------------------
extern "C" void kernel_launch(void* const* d_in, const int* in_sizes, int n_in,
                              void* d_out, int out_size, void* d_ws,
                              size_t ws_size, hipStream_t stream) {
    const float* x  = (const float*)d_in[0];
    const float* Wq = (const float*)d_in[1];
    const float* bq = (const float*)d_in[2];
    const float* Wk = (const float*)d_in[3];
    const float* bk = (const float*)d_in[4];
    const float* Wv = (const float*)d_in[5];
    const float* bv = (const float*)d_in[6];
    const float* Wo = (const float*)d_in[7];
    const float* bo = (const float*)d_in[8];
    float* out = (float*)d_out;

    char* w = (char*)d_ws;
    const size_t SZ_X = (size_t)MTOT * EDIM * 2;   // 8MB
    const size_t SZ_W = (size_t)EDIM * EDIM * 2;   // 2MB
    _Float16* x_h  = (_Float16*)(w);            w += SZ_X;
    _Float16* Wq_h = (_Float16*)(w);            w += SZ_W;
    _Float16* Wk_h = (_Float16*)(w);            w += SZ_W;
    _Float16* Wv_h = (_Float16*)(w);            w += SZ_W;
    _Float16* Wo_h = (_Float16*)(w);            w += SZ_W;
    _Float16* Q_h  = (_Float16*)(w);            w += SZ_X;
    _Float16* K_h  = (_Float16*)(w);            w += SZ_X;
    _Float16* VT_h = (_Float16*)(w);            w += SZ_X;
    _Float16* Ctx  = (_Float16*)(w);            w += SZ_X;

    // 1) converts
    {
        int n = MTOT * EDIM;
        mhsa_cvt_f16<<<(n + 255) / 256, 256, 0, stream>>>(x, x_h, n);
        n = EDIM * EDIM;
        mhsa_cvt_f16<<<(n + 255) / 256, 256, 0, stream>>>(Wq, Wq_h, n);
        mhsa_cvt_f16<<<(n + 255) / 256, 256, 0, stream>>>(Wk, Wk_h, n);
        mhsa_cvt_f16<<<(n + 255) / 256, 256, 0, stream>>>(Wv, Wv_h, n);
        mhsa_cvt_f16<<<(n + 255) / 256, 256, 0, stream>>>(Wo, Wo_h, n);
    }

    // 2) Q/K/V projections
    dim3 gg(EDIM / 128, MTOT / 128);   // 8 x 32 = 256 blocks
    mhsa_gemm_xwt<<<gg, 256, 0, stream>>>(x_h, Wq_h, bq, (void*)Q_h,
                                          MTOT, EDIM, EDIM, 0);
    mhsa_gemm_xwt<<<gg, 256, 0, stream>>>(x_h, Wk_h, bk, (void*)K_h,
                                          MTOT, EDIM, EDIM, 0);
    mhsa_gemm_xwt<<<gg, 256, 0, stream>>>(x_h, Wv_h, bv, (void*)VT_h,
                                          MTOT, EDIM, EDIM, 1);

    // 3) attention
    dim3 ga(BATCH * NHEADS, SEQ / RBLK);  // 64 x 8 = 512 blocks
    size_t smem = (size_t)RBLK * PSTR * 2;  // ~258KB (<=320KB WGP LDS)
    mhsa_attn<<<ga, 256, smem, stream>>>(Q_h, K_h, VT_h, Ctx);

    // 4) output projection -> f32 d_out
    mhsa_gemm_xwt<<<gg, 256, 0, stream>>>(Ctx, Wo_h, bo, (void*)out,
                                          MTOT, EDIM, EDIM, 2);
}
